// GraphGNNModel_5007931867540
// MI455X (gfx1250) — matrix operand order repaired and checked by reading, not compiled, not run on hardware
//
#include <hip/hip_runtime.h>
#include <math.h>

#define NN      50000        // nodes
#define NE      800000       // edges (before self loops)
#define NG      256          // graphs
#define CIN     128
#define CH0     64           // layer0 per-head out
#define CH1     128          // layer1 per-head out
#define H       2            // heads
#define HC0     (H*CH0)      // 128
#define HC1     (H*CH1)      // 256
#define NEG_SLOPE 0.2f
#define BN_EPS  1e-5f
#define ETOT    (NE + NN)    // with self loops

typedef __attribute__((ext_vector_type(16))) _Float16 v16h;
typedef __attribute__((ext_vector_type(8)))  float    v8f;

static inline int ceil_div(long a, long b) { return (int)((a + b - 1) / b); }

// ---------------- utility ----------------
__global__ void k_fill_u32(unsigned* p, unsigned v, long n) {
  long i = (long)blockIdx.x * blockDim.x + threadIdx.x;
  long st = (long)gridDim.x * blockDim.x;
  for (; i < n; i += st) p[i] = v;
}

__global__ void k_f32_to_f16(const float* __restrict__ in, _Float16* __restrict__ out, long n) {
  long i = (long)blockIdx.x * blockDim.x + threadIdx.x;
  long st = (long)gridDim.x * blockDim.x;
  for (; i < n; i += st) out[i] = (_Float16)in[i];
}

// Transpose weights to f16 column-major: Bt[n*K + k] = W[k*N + n]
__global__ void k_w_f16_t(const float* __restrict__ W, _Float16* __restrict__ Bt, int K, int N) {
  long i = (long)blockIdx.x * blockDim.x + threadIdx.x;
  long tot = (long)K * N;
  long st = (long)gridDim.x * blockDim.x;
  for (; i < tot; i += st) {
    int n = (int)(i / K), k = (int)(i % K);
    Bt[i] = (_Float16)W[(size_t)k * N + n];
  }
}

// ---------------- WMMA GEMM: C[M,N] = A[M,K] x Bt[N,K]^T, f16 in / f32 out ----------
// One wave computes a 16x64 output strip: 1 A-fragment feeds 4 WMMAs per K-step.
__global__ __launch_bounds__(256)
void k_wmma_gemm4(const _Float16* __restrict__ A, const _Float16* __restrict__ Bt,
                  float* __restrict__ C, int M, int N, int K) {
  int lane = threadIdx.x & 31;
  int wave = threadIdx.x >> 5;
  int groupsN = N >> 6;                      // 64-wide column groups
  int numTiles = (M >> 4) * groupsN;
  int tile = blockIdx.x * 8 + wave;
  if (tile >= numTiles) return;              // wave-uniform: EXEC stays all-ones
  int tm = tile / groupsN, tg = tile - tm * groupsN;
  int m0 = tm << 4, n0 = tg << 6;
  int r16  = lane & 15;                      // row (A) / col (B,C)
  int half = lane >> 4;                      // K-half selector
  v8f acc[4] = {{}, {}, {}, {}};
  const _Float16* arow = A + (size_t)(m0 + r16) * K;
  for (int k0 = 0; k0 < K; k0 += 32) {
    v16h a;
    const _Float16* ap = arow + k0 + half * 8;   // two contiguous b128 chunks
#pragma unroll
    for (int e = 0; e < 8; ++e) a[e] = ap[e];
#pragma unroll
    for (int e = 0; e < 8; ++e) a[8 + e] = ap[16 + e];
    __builtin_prefetch(ap + 32, 0, 0);           // global_prefetch_b8: next K-step of A
#pragma unroll
    for (int t = 0; t < 4; ++t) {
      v16h b;
      // lane col = r16; two 8-wide K chunks (k0+half*8, k0+16+half*8) from transposed weights
      const _Float16* bp = Bt + (size_t)(n0 + t * 16 + r16) * K + k0 + half * 8;
#pragma unroll
      for (int e = 0; e < 16; ++e) b[e] = bp[(e & 7) + ((e >> 3) << 4)];
      acc[t] = __builtin_amdgcn_wmma_f32_16x16x32_f16(false, a, false, b,
                                                      (short)0, acc[t], false, false);
    }
  }
  // C/D layout: VGPR r -> row r + half*8, col r16
#pragma unroll
  for (int t = 0; t < 4; ++t)
#pragma unroll
    for (int r = 0; r < 8; ++r)
      C[(size_t)(m0 + r + half * 8) * N + n0 + t * 16 + r16] = acc[t][r];
}

// ---------------- attention scores a_s/a_d per (node, head) ----------------
__global__ void k_att_scores(const float* __restrict__ h, const float* __restrict__ att_s,
                             const float* __restrict__ att_d, float* __restrict__ a_s,
                             float* __restrict__ a_d, int C) {
  int n = blockIdx.x * blockDim.x + threadIdx.x;
  if (n >= NN) return;
  for (int hd = 0; hd < H; ++hd) {
    const float* hp = h + (size_t)n * H * C + hd * C;
    float ss = 0.f, sd = 0.f;
    for (int c = 0; c < C; ++c) { float v = hp[c]; ss += v * att_s[hd * C + c]; sd += v * att_d[hd * C + c]; }
    a_s[n * H + hd] = ss;
    a_d[n * H + hd] = sd;
  }
}

// order-preserving float<->uint encoding for atomicMax on floats
__device__ __forceinline__ unsigned fenc(float x) {
  unsigned u = __float_as_uint(x);
  return (u & 0x80000000u) ? ~u : (u | 0x80000000u);
}
__device__ __forceinline__ float fdec(unsigned u) {
  return (u & 0x80000000u) ? __uint_as_float(u & 0x7FFFFFFFu) : __uint_as_float(~u);
}

__device__ __forceinline__ void edge_sd(const int* __restrict__ ei, long idx,
                                        int& s, int& d, int& hd) {
  long e = idx >> 1; hd = (int)(idx & 1);
  if (e < NE) { s = ei[e]; d = ei[NE + e]; }
  else        { s = d = (int)(e - NE); }     // self-loop
}

__global__ void k_edge_max(const int* __restrict__ ei, const float* __restrict__ a_s,
                           const float* __restrict__ a_d, unsigned* __restrict__ menc) {
  long idx = (long)blockIdx.x * blockDim.x + threadIdx.x;
  if (idx >= (long)ETOT * H) return;
  int s, d, hd; edge_sd(ei, idx, s, d, hd);
  float e = a_s[s * H + hd] + a_d[d * H + hd];
  e = e > 0.f ? e : NEG_SLOPE * e;
  atomicMax(&menc[d * H + hd], fenc(e));
}

__global__ void k_edge_sum(const int* __restrict__ ei, const float* __restrict__ a_s,
                           const float* __restrict__ a_d, const unsigned* __restrict__ menc,
                           float* __restrict__ den) {
  long idx = (long)blockIdx.x * blockDim.x + threadIdx.x;
  if (idx >= (long)ETOT * H) return;
  int s, d, hd; edge_sd(ei, idx, s, d, hd);
  float e = a_s[s * H + hd] + a_d[d * H + hd];
  e = e > 0.f ? e : NEG_SLOPE * e;
  atomicAdd(&den[d * H + hd], expf(e - fdec(menc[d * H + hd])));
}

__global__ void k_edge_agg(const int* __restrict__ ei, const float* __restrict__ a_s,
                           const float* __restrict__ a_d, const unsigned* __restrict__ menc,
                           const float* __restrict__ den, const float* __restrict__ h,
                           float* __restrict__ out, int C) {
  long idx = (long)blockIdx.x * blockDim.x + threadIdx.x;
  if (idx >= (long)ETOT * H) return;
  int s, d, hd; edge_sd(ei, idx, s, d, hd);
  float e = a_s[s * H + hd] + a_d[d * H + hd];
  e = e > 0.f ? e : NEG_SLOPE * e;
  float ex = expf(e - fdec(menc[d * H + hd]));
  float dn = den[d * H + hd]; dn = dn > 1e-16f ? dn : 1e-16f;
  float alpha = ex / dn;
  const float* hs = h + (size_t)s * H * C + hd * C;
  float* od = out + (size_t)d * H * C + hd * C;
  for (int c = 0; c < C; ++c) atomicAdd(&od[c], alpha * hs[c]);
}

// ---------------- per-channel BN stats over nodes ----------------
__global__ void k_col_stats(const float* __restrict__ in, const float* __restrict__ bias,
                            float* __restrict__ mu, float* __restrict__ rsig, int HC) {
  int c = blockIdx.x;
  __shared__ float sh[256], sh2[256];
  float b = bias[c], s = 0.f, s2 = 0.f;
  for (int r = threadIdx.x; r < NN; r += 256) {
    float v = in[(size_t)r * HC + c] + b;
    s += v; s2 += v * v;
  }
  sh[threadIdx.x] = s; sh2[threadIdx.x] = s2;
  __syncthreads();
  for (int o = 128; o > 0; o >>= 1) {
    if (threadIdx.x < (unsigned)o) { sh[threadIdx.x] += sh[threadIdx.x + o]; sh2[threadIdx.x] += sh2[threadIdx.x + o]; }
    __syncthreads();
  }
  if (threadIdx.x == 0) {
    float m = sh[0] / (float)NN;
    float var = sh2[0] / (float)NN - m * m;
    mu[c] = m;
    rsig[c] = rsqrtf(var + BN_EPS);
  }
}

__global__ void k_bn_relu_f16(const float* __restrict__ in, const float* __restrict__ bias,
                              const float* __restrict__ mu, const float* __restrict__ rsig,
                              const float* __restrict__ g, const float* __restrict__ be,
                              _Float16* __restrict__ out, int HC) {
  long i = (long)blockIdx.x * blockDim.x + threadIdx.x;
  long n = (long)NN * HC;
  long st = (long)gridDim.x * blockDim.x;
  for (; i < n; i += st) {
    int c = (int)(i % HC);
    float v = (in[i] + bias[c] - mu[c]) * rsig[c] * g[c] + be[c];
    out[i] = (_Float16)(v > 0.f ? v : 0.f);
  }
}

__global__ void k_bn_relu_f32(float* __restrict__ io, const float* __restrict__ bias,
                              const float* __restrict__ mu, const float* __restrict__ rsig,
                              const float* __restrict__ g, const float* __restrict__ be, int HC) {
  long i = (long)blockIdx.x * blockDim.x + threadIdx.x;
  long n = (long)NN * HC;
  long st = (long)gridDim.x * blockDim.x;
  for (; i < n; i += st) {
    int c = (int)(i % HC);
    float v = (io[i] + bias[c] - mu[c]) * rsig[c] * g[c] + be[c];
    io[i] = v > 0.f ? v : 0.f;
  }
}

// ---------------- mean pool + head ----------------
__global__ void k_pool_count(const int* __restrict__ batch, float* __restrict__ cnt) {
  int n = blockIdx.x * blockDim.x + threadIdx.x;
  if (n < NN) atomicAdd(&cnt[batch[n]], 1.0f);
}
__global__ void k_pool_sum(const float* __restrict__ h, const int* __restrict__ batch,
                           float* __restrict__ sums, int HC) {
  long i = (long)blockIdx.x * blockDim.x + threadIdx.x;
  long n = (long)NN * HC;
  long st = (long)gridDim.x * blockDim.x;
  for (; i < n; i += st) {
    int node = (int)(i / HC), c = (int)(i % HC);
    atomicAdd(&sums[(size_t)batch[node] * HC + c], h[i]);
  }
}
__global__ void k_head1(const float* __restrict__ sums, const float* __restrict__ cnt,
                        const float* __restrict__ W, const float* __restrict__ b,
                        float* __restrict__ hid) {
  int idx = blockIdx.x * blockDim.x + threadIdx.x;      // NG*256
  int g = idx >> 8, j = idx & 255;
  float inv = 1.0f / fmaxf(cnt[g], 1.0f);
  float acc = b[j];
  for (int k = 0; k < 256; ++k) acc += sums[g * 256 + k] * inv * W[k * 256 + j];
  hid[idx] = fmaxf(acc, 0.f);
}
__global__ void k_head2(const float* __restrict__ hid, const float* __restrict__ W,
                        const float* __restrict__ b, float* __restrict__ out) {
  int g = blockIdx.x * blockDim.x + threadIdx.x;
  if (g >= NG) return;
  float acc = b[0];
  for (int j = 0; j < 256; ++j) acc += hid[g * 256 + j] * W[j];
  out[g] = acc;
}

// ---------------- launch ----------------
extern "C" void kernel_launch(void* const* d_in, const int* in_sizes, int n_in,
                              void* d_out, int out_size, void* d_ws, size_t ws_size,
                              hipStream_t stream) {
  (void)in_sizes; (void)n_in; (void)out_size; (void)ws_size;
  const float* x        = (const float*)d_in[0];
  const int*   ei       = (const int*)  d_in[1];
  const int*   batch    = (const int*)  d_in[2];
  const float* W0       = (const float*)d_in[3];
  const float* as0      = (const float*)d_in[4];
  const float* ad0      = (const float*)d_in[5];
  const float* b0       = (const float*)d_in[6];
  const float* g0       = (const float*)d_in[7];
  const float* be0      = (const float*)d_in[8];
  const float* W1       = (const float*)d_in[9];
  const float* as1      = (const float*)d_in[10];
  const float* ad1      = (const float*)d_in[11];
  const float* b1       = (const float*)d_in[12];
  const float* g1       = (const float*)d_in[13];
  const float* be1      = (const float*)d_in[14];
  const float* Wh1      = (const float*)d_in[15];
  const float* bh1      = (const float*)d_in[16];
  const float* Wh2      = (const float*)d_in[17];
  const float* bh2      = (const float*)d_in[18];
  float* out = (float*)d_out;

  // workspace carve-out (256B aligned)
  size_t off = 0;
  auto carve = [&](size_t bytes) {
    void* p = (char*)d_ws + off;
    off += (bytes + 255) & ~(size_t)255;
    return p;
  };
  _Float16* xh   = (_Float16*)carve((size_t)NN * CIN * 2);
  _Float16* W0t  = (_Float16*)carve((size_t)CIN * HC0 * 2);   // [HC0][CIN] f16
  _Float16* W1t  = (_Float16*)carve((size_t)HC0 * HC1 * 2);   // [HC1][HC0] f16
  float*    h0   = (float*)   carve((size_t)NN * HC0 * 4);
  float*    agg0 = (float*)   carve((size_t)NN * HC0 * 4);
  _Float16* hn0  = (_Float16*)carve((size_t)NN * HC0 * 2);
  float*    h1   = (float*)   carve((size_t)NN * HC1 * 4);
  float*    agg1 = (float*)   carve((size_t)NN * HC1 * 4);
  float*    a_s  = (float*)   carve((size_t)NN * H * 4);
  float*    a_d  = (float*)   carve((size_t)NN * H * 4);
  unsigned* menc = (unsigned*)carve((size_t)NN * H * 4);
  float*    den  = (float*)   carve((size_t)NN * H * 4);
  float*    mu   = (float*)   carve(256 * 4);
  float*    rsig = (float*)   carve(256 * 4);
  float*    psum = (float*)   carve((size_t)NG * HC1 * 4);
  float*    pcnt = (float*)   carve(NG * 4);
  float*    hid  = (float*)   carve((size_t)NG * 256 * 4);

  const int T = 256;
  const long eWork = (long)ETOT * H;
  const int eBlk = ceil_div(eWork, T);

  // conversions / weight transposes
  k_f32_to_f16<<<2048, T, 0, stream>>>(x, xh, (long)NN * CIN);
  k_w_f16_t<<<ceil_div((long)CIN * HC0, T), T, 0, stream>>>(W0, W0t, CIN, HC0);
  k_w_f16_t<<<ceil_div((long)HC0 * HC1, T), T, 0, stream>>>(W1, W1t, HC0, HC1);

  // ---- layer 0 ----
  {
    int tiles = (NN / 16) * (HC0 / 64);
    k_wmma_gemm4<<<ceil_div(tiles, 8), T, 0, stream>>>(xh, W0t, h0, NN, HC0, CIN);
  }
  k_att_scores<<<ceil_div(NN, T), T, 0, stream>>>(h0, as0, ad0, a_s, a_d, CH0);
  k_fill_u32<<<1024, T, 0, stream>>>(menc, 0u, (long)NN * H);
  k_fill_u32<<<1024, T, 0, stream>>>((unsigned*)den, 0u, (long)NN * H);
  k_fill_u32<<<4096, T, 0, stream>>>((unsigned*)agg0, 0u, (long)NN * HC0);
  k_edge_max<<<eBlk, T, 0, stream>>>(ei, a_s, a_d, menc);
  k_edge_sum<<<eBlk, T, 0, stream>>>(ei, a_s, a_d, menc, den);
  k_edge_agg<<<eBlk, T, 0, stream>>>(ei, a_s, a_d, menc, den, h0, agg0, CH0);
  k_col_stats<<<HC0, T, 0, stream>>>(agg0, b0, mu, rsig, HC0);
  k_bn_relu_f16<<<4096, T, 0, stream>>>(agg0, b0, mu, rsig, g0, be0, hn0, HC0);

  // ---- layer 1 ----
  {
    int tiles = (NN / 16) * (HC1 / 64);
    k_wmma_gemm4<<<ceil_div(tiles, 8), T, 0, stream>>>(hn0, W1t, h1, NN, HC1, HC0);
  }
  k_att_scores<<<ceil_div(NN, T), T, 0, stream>>>(h1, as1, ad1, a_s, a_d, CH1);
  k_fill_u32<<<1024, T, 0, stream>>>(menc, 0u, (long)NN * H);
  k_fill_u32<<<1024, T, 0, stream>>>((unsigned*)den, 0u, (long)NN * H);
  k_fill_u32<<<4096, T, 0, stream>>>((unsigned*)agg1, 0u, (long)NN * HC1);
  k_edge_max<<<eBlk, T, 0, stream>>>(ei, a_s, a_d, menc);
  k_edge_sum<<<eBlk, T, 0, stream>>>(ei, a_s, a_d, menc, den);
  k_edge_agg<<<eBlk, T, 0, stream>>>(ei, a_s, a_d, menc, den, h1, agg1, CH1);
  k_col_stats<<<HC1, T, 0, stream>>>(agg1, b1, mu, rsig, HC1);
  k_bn_relu_f32<<<4096, T, 0, stream>>>(agg1, b1, mu, rsig, g1, be1, HC1);

  // ---- pool + head ----
  k_fill_u32<<<ceil_div((long)NG * HC1, T), T, 0, stream>>>((unsigned*)psum, 0u, (long)NG * HC1);
  k_fill_u32<<<1, T, 0, stream>>>((unsigned*)pcnt, 0u, NG);
  k_pool_count<<<ceil_div(NN, T), T, 0, stream>>>(batch, pcnt);
  k_pool_sum<<<4096, T, 0, stream>>>(agg1, batch, psum, HC1);
  k_head1<<<NG, T, 0, stream>>>(psum, pcnt, Wh1, bh1, hid);
  k_head2<<<1, T, 0, stream>>>(hid, Wh2, bh2, out);
}
